// NATFLayer_79568564126297
// MI455X (gfx1250) — compile-verified, matrix operand order checked by reading, not run on hardware
//
#include <hip/hip_runtime.h>
#include <hip/hip_bf16.h>

// ---------------------------------------------------------------------------
// Types for CDNA5 WMMA (wave32): v16bf A/B fragments, v8f accumulator.
// ---------------------------------------------------------------------------
typedef __bf16 bf16x8  __attribute__((ext_vector_type(8)));
typedef __bf16 bf16x16 __attribute__((ext_vector_type(16)));
typedef float  f32x8   __attribute__((ext_vector_type(8)));

#define T_SEQ 2048
#define C_DIM 512
#define N_HEAD 8
#define HD 64
#define WIN 63
#define LEFT 31

// LDS tile geometry: k-step of 32 bf16 per row, row stride padded to 40
// elements (80 B = 20 words) -> ds_load_b128 across 16 lanes hits 16 distinct
// bank groups (20*r mod 64 distinct for r = 0..15): conflict-free.
#define KSTEP   32
#define LDS_LDA 40   // elements per row (32 data + 8 pad)

__device__ inline float gelu_exact(float x) {
    return 0.5f * x * (1.0f + erff(x * 0.70710678118654752f));
}

// LDS byte offset of a generic (__shared__) pointer: the shared aperture maps
// addr[31:0] onto the workgroup's LDS allocation.
__device__ inline unsigned lds_off(const void* p) {
    return (unsigned)(size_t)p;
}

// CDNA5 async global -> LDS copy, 16 bytes per lane. Tracked by ASYNCcnt.
__device__ inline void async_load_b128(unsigned lds_addr, const void* gaddr) {
    asm volatile("global_load_async_to_lds_b128 %0, %1, off"
                 :: "v"(lds_addr), "v"(gaddr) : "memory");
}
__device__ inline void wait_asynccnt0() {
    asm volatile("s_wait_asynccnt 0x0" ::: "memory");
}

// Fragment loader from an LDS row (row-major, K-contiguous). Lane l holds row
// (l&15); K-runs [koff, koff+8) and [koff+16, koff+24), koff = (l>>4)*8.
__device__ inline bf16x16 load_frag_lds(const __bf16* rowp, int koff) {
    union { bf16x16 v; bf16x8 h[2]; } u;
    u.h[0] = *reinterpret_cast<const bf16x8*>(rowp + koff);
    u.h[1] = *reinterpret_cast<const bf16x8*>(rowp + koff + 16);
    return u.v;
}

// ---------------------------------------------------------------------------
// bf16 WMMA GEMM:  out[M,N] = A[M,K] * B[N,K]^T  (+bias, epilogue by MODE)
//   MODE 0: f32 out = acc + bias
//   MODE 1: f32 out = acc + bias + resid
//   MODE 2: bf16 out = gelu(acc + bias)
// Block: 256 threads (8 waves), 128(M) x 256(N) tile; wave (wm,wn) in a 2x4
// grid owns a 64x64 sub-tile = 16 accumulators of 16x16. A/B tiles for each
// 32-wide k-step are staged in a DOUBLE-BUFFERED LDS pair via
// global_load_async_to_lds_b128: slab k+1 streams in (ASYNCcnt-tracked)
// while the 16 WMMAs of slab k execute.
// ---------------------------------------------------------------------------
template <int MODE>
__global__ __launch_bounds__(256)
void gemm_bf16_wmma(const __bf16* __restrict__ A, const __bf16* __restrict__ B,
                    const float* __restrict__ bias, const float* __restrict__ resid,
                    void* __restrict__ outv, int M, int N, int K) {
    __shared__ __bf16 As[2][128 * LDS_LDA];   // 2 x 10,240 B
    __shared__ __bf16 Bs[2][256 * LDS_LDA];   // 2 x 20,480 B

    const int tid  = threadIdx.x;
    const int lane = tid & 31;
    const int wave = tid >> 5;
    const int wm = wave >> 2;              // 0..1
    const int wn = wave & 3;               // 0..3
    const int mblk = blockIdx.y * 128;
    const int nblk = blockIdx.x * 256;
    const int m0 = mblk + wm * 64;
    const int n0 = nblk + wn * 64;
    const int r    = lane & 15;
    const int koff = (lane >> 4) * 8;

    f32x8 acc[16] = {};

    // Async-copy assignments: tiles split into 16-byte chunks
    // (4 chunks of 8 bf16 per 32-element row).
    //   A tile: 128 rows * 4 chunks = 512  -> 2 chunks/thread
    //   B tile: 256 rows * 4 chunks = 1024 -> 4 chunks/thread
    const unsigned aBufBytes = (unsigned)sizeof(As[0]);
    const unsigned bBufBytes = (unsigned)sizeof(Bs[0]);
    const unsigned asBase = lds_off(As);
    const unsigned bsBase = lds_off(Bs);

    int aRow[2], aCc[2]; unsigned aLds[2];
#pragma unroll
    for (int i = 0; i < 2; ++i) {
        const int c = tid + i * 256;
        aRow[i] = c >> 2; aCc[i] = c & 3;
        aLds[i] = asBase + (unsigned)(aRow[i] * (LDS_LDA * 2) + aCc[i] * 16);
    }
    int bRow[4], bCc[4]; unsigned bLds[4];
#pragma unroll
    for (int i = 0; i < 4; ++i) {
        const int c = tid + i * 256;
        bRow[i] = c >> 2; bCc[i] = c & 3;
        bLds[i] = bsBase + (unsigned)(bRow[i] * (LDS_LDA * 2) + bCc[i] * 16);
    }

    // Stage one 32-wide k-slab into LDS buffer `buf` (async).
    auto stage = [&](int buf, int k) {
#pragma unroll
        for (int i = 0; i < 2; ++i)
            async_load_b128(aLds[i] + (buf ? aBufBytes : 0),
                A + (size_t)(mblk + aRow[i]) * K + k + aCc[i] * 8);
#pragma unroll
        for (int i = 0; i < 4; ++i)
            async_load_b128(bLds[i] + (buf ? bBufBytes : 0),
                B + (size_t)(nblk + bRow[i]) * K + k + bCc[i] * 8);
    };

    // Prologue: fill buffer 0 with the first slab.
    stage(0, 0);
    int cur = 0;

    for (int k = 0; k < K; k += KSTEP) {
        wait_asynccnt0();  // my async writes into buf `cur` landed
        __syncthreads();   // everyone's landed; prior reads of buf `cur^1` done

        // Stream the next slab into the other buffer while we compute.
        if (k + KSTEP < K) stage(cur ^ 1, k + KSTEP);

        // Warm the cache hierarchy two slabs ahead.
        if (k + 2 * KSTEP < K) {
#pragma unroll
            for (int i = 0; i < 2; ++i)
                __builtin_prefetch(A + (size_t)(mblk + aRow[i]) * K + k + 2 * KSTEP, 0, 3);
#pragma unroll
            for (int i = 0; i < 4; ++i)
                __builtin_prefetch(B + (size_t)(nblk + bRow[i]) * K + k + 2 * KSTEP, 0, 3);
        }

        const __bf16* aBase = As[cur];
        const __bf16* bBase = Bs[cur];
        bf16x16 af[4];
#pragma unroll
        for (int i = 0; i < 4; ++i)
            af[i] = load_frag_lds(aBase + (wm * 64 + i * 16 + r) * LDS_LDA, koff);
#pragma unroll
        for (int j = 0; j < 4; ++j) {
            const bf16x16 bfr =
                load_frag_lds(bBase + (wn * 64 + j * 16 + r) * LDS_LDA, koff);
#pragma unroll
            for (int i = 0; i < 4; ++i)
                acc[i * 4 + j] = __builtin_amdgcn_wmma_f32_16x16x32_bf16(
                    false, af[i], false, bfr, (short)0, acc[i * 4 + j],
                    false, false);
        }
        cur ^= 1;
    }

    // Epilogue. C layout: lane l -> col (l&15), VGPR rr -> row rr + (l>>4)*8.
    const int rbase = (lane >> 4) * 8;
#pragma unroll
    for (int i = 0; i < 4; ++i) {
#pragma unroll
        for (int j = 0; j < 4; ++j) {
            const int nn = n0 + j * 16 + (lane & 15);
            const float bv = bias[nn];
#pragma unroll
            for (int rr = 0; rr < 8; ++rr) {
                const int mm = m0 + i * 16 + rbase + rr;
                float v = acc[i * 4 + j][rr] + bv;
                const size_t idx = (size_t)mm * N + nn;
                if (MODE == 0) {
                    ((float*)outv)[idx] = v;
                } else if (MODE == 1) {
                    ((float*)outv)[idx] = v + resid[idx];
                } else {
                    ((__bf16*)outv)[idx] = (__bf16)gelu_exact(v);
                }
            }
        }
    }
}

// ---------------------------------------------------------------------------
// LayerNorm over C=512, one block of 128 threads per row, output bf16.
// ---------------------------------------------------------------------------
__global__ __launch_bounds__(128)
void layernorm512_bf16(const float* __restrict__ x, const float* __restrict__ w,
                       const float* __restrict__ b, __bf16* __restrict__ out) {
    __shared__ float red[4];
    const int row = blockIdx.x;
    const int tid = threadIdx.x;
    const float* xr = x + (size_t)row * C_DIM;
    float v[4];
    float s = 0.f;
#pragma unroll
    for (int i = 0; i < 4; ++i) { v[i] = xr[tid + i * 128]; s += v[i]; }
#pragma unroll
    for (int o = 16; o > 0; o >>= 1) s += __shfl_xor(s, o, 32);
    if ((tid & 31) == 0) red[tid >> 5] = s;
    __syncthreads();
    const float mean = (red[0] + red[1] + red[2] + red[3]) * (1.f / 512.f);
    __syncthreads();
    float sq = 0.f;
#pragma unroll
    for (int i = 0; i < 4; ++i) { float d = v[i] - mean; sq += d * d; }
#pragma unroll
    for (int o = 16; o > 0; o >>= 1) sq += __shfl_xor(sq, o, 32);
    if ((tid & 31) == 0) red[tid >> 5] = sq;
    __syncthreads();
    const float var  = (red[0] + red[1] + red[2] + red[3]) * (1.f / 512.f);
    const float rstd = rsqrtf(var + 1e-5f);
    __bf16* orow = out + (size_t)row * C_DIM;
#pragma unroll
    for (int i = 0; i < 4; ++i) {
        const int c = tid + i * 128;
        orow[c] = (__bf16)((v[i] - mean) * rstd * w[c] + b[c]);
    }
}

// ---------------------------------------------------------------------------
// f32 -> bf16 elementwise (weight conversion)
// ---------------------------------------------------------------------------
__global__ __launch_bounds__(256)
void f32_to_bf16(const float* __restrict__ in, __bf16* __restrict__ out, int n) {
    const int i = blockIdx.x * 256 + threadIdx.x;
    if (i < n) out[i] = (__bf16)in[i];
}

// ---------------------------------------------------------------------------
// Windowed attention. One block per (n,t) row; wave w handles head w (8 waves).
// qkv: f32, row layout [q(512) | k(512) | v(512)], head h at offset h*64.
// Output: bf16 (feeds the proj WMMA GEMM).
// ---------------------------------------------------------------------------
__global__ __launch_bounds__(256)
void window_attn(const float* __restrict__ qkv, const float* __restrict__ rel_bias,
                 __bf16* __restrict__ out) {
    __shared__ float attn_s[N_HEAD][64];
    const int lane = threadIdx.x & 31;
    const int head = threadIdx.x >> 5;
    const int m = blockIdx.x;        // n*T + t
    const int t = m & (T_SEQ - 1);
    const int nrow0 = m - t;         // n*T
    const float* q = qkv + (size_t)m * (3 * C_DIM) + head * HD;

    float s[2];
#pragma unroll
    for (int rep = 0; rep < 2; ++rep) {
        const int w = lane + rep * 32;
        float sc = -1e30f;           // w==63 excluded from softmax
        if (w < WIN) {
            const int pos = t - LEFT + w;
            if (pos >= 0 && pos < T_SEQ) {
                const float* kk = qkv + (size_t)(nrow0 + pos) * (3 * C_DIM)
                                  + C_DIM + head * HD;
                float dot = 0.f;
#pragma unroll 8
                for (int d = 0; d < HD; ++d) dot += q[d] * kk[d];
                sc = dot * 0.125f + rel_bias[head * WIN + w];
            } else {
                sc = rel_bias[head * WIN + w] - 100.0f;
            }
        }
        s[rep] = sc;
    }
    float mx = fmaxf(s[0], s[1]);
#pragma unroll
    for (int o = 16; o > 0; o >>= 1) mx = fmaxf(mx, __shfl_xor(mx, o, 32));
    const float e0 = __expf(s[0] - mx);
    const float e1 = __expf(s[1] - mx);   // lane 31 -> exp(-huge) = 0
    float sum = e0 + e1;
#pragma unroll
    for (int o = 16; o > 0; o >>= 1) sum += __shfl_xor(sum, o, 32);
    const float inv = 1.f / sum;
    attn_s[head][lane]      = e0 * inv;
    attn_s[head][lane + 32] = e1 * inv;
    __syncthreads();

    float a0 = 0.f, a1 = 0.f;
    for (int w = 0; w < WIN; ++w) {
        const int pos = t - LEFT + w;
        if (pos < 0 || pos >= T_SEQ) continue;
        const float a = attn_s[head][w];
        const float* vv = qkv + (size_t)(nrow0 + pos) * (3 * C_DIM)
                          + 2 * C_DIM + head * HD;
        a0 += a * vv[lane];
        a1 += a * vv[lane + 32];
    }
    __bf16* orow = out + (size_t)m * C_DIM + head * HD;
    orow[lane]      = (__bf16)a0;
    orow[lane + 32] = (__bf16)a1;
}

// ---------------------------------------------------------------------------
// Host-side launch orchestration.
// ---------------------------------------------------------------------------
extern "C" void kernel_launch(void* const* d_in, const int* in_sizes, int n_in,
                              void* d_out, int out_size, void* d_ws, size_t ws_size,
                              hipStream_t stream) {
    const float* x       = (const float*)d_in[0];
    const float* norm1_w = (const float*)d_in[1];
    const float* norm1_b = (const float*)d_in[2];
    const float* qkv_w   = (const float*)d_in[3];
    const float* qkv_b   = (const float*)d_in[4];
    const float* rel_b   = (const float*)d_in[5];
    const float* proj_w  = (const float*)d_in[6];
    const float* proj_b  = (const float*)d_in[7];
    const float* norm2_w = (const float*)d_in[8];
    const float* norm2_b = (const float*)d_in[9];
    const float* fc1_w   = (const float*)d_in[10];
    const float* fc1_b   = (const float*)d_in[11];
    const float* fc2_w   = (const float*)d_in[12];
    const float* fc2_b   = (const float*)d_in[13];
    float* out = (float*)d_out;

    const int M = 2 * T_SEQ;            // 4096 rows
    const int C = C_DIM;                // 512
    const int HID = 4 * C;              // 2048

    // Workspace carve-out (256B aligned).
    char* ws = (char*)d_ws;
    size_t off = 0;
    auto carve = [&](size_t bytes) -> void* {
        void* p = ws + off;
        off += (bytes + 255) & ~(size_t)255;
        return p;
    };
    __bf16* h_bf    = (__bf16*)carve((size_t)M * C * 2);          // LN1 out / LN2 out
    __bf16* qkvw_bf = (__bf16*)carve((size_t)3 * C * C * 2);
    __bf16* projw_bf= (__bf16*)carve((size_t)C * C * 2);
    __bf16* fc1w_bf = (__bf16*)carve((size_t)HID * C * 2);
    __bf16* fc2w_bf = (__bf16*)carve((size_t)C * HID * 2);
    float*  qkv_f32 = (float*)carve((size_t)M * 3 * C * 4);
    __bf16* attn_bf = (__bf16*)carve((size_t)M * C * 2);
    float*  x2_f32  = (float*)carve((size_t)M * C * 4);
    __bf16* m_bf    = (__bf16*)carve((size_t)M * HID * 2);
    (void)ws_size; (void)in_sizes; (void)n_in; (void)out_size;

    // 1) Weight conversion f32 -> bf16.
    f32_to_bf16<<<(3 * C * C + 255) / 256, 256, 0, stream>>>(qkv_w, qkvw_bf, 3 * C * C);
    f32_to_bf16<<<(C * C + 255) / 256, 256, 0, stream>>>(proj_w, projw_bf, C * C);
    f32_to_bf16<<<(HID * C + 255) / 256, 256, 0, stream>>>(fc1_w, fc1w_bf, HID * C);
    f32_to_bf16<<<(C * HID + 255) / 256, 256, 0, stream>>>(fc2_w, fc2w_bf, C * HID);

    // 2) LN1 -> bf16 activations.
    layernorm512_bf16<<<M, 128, 0, stream>>>(x, norm1_w, norm1_b, h_bf);

    // 3) QKV GEMM: (4096x512)*(1536x512)^T -> f32.
    gemm_bf16_wmma<0><<<dim3((3 * C) / 256, M / 128), 256, 0, stream>>>(
        h_bf, qkvw_bf, qkv_b, nullptr, qkv_f32, M, 3 * C, C);

    // 4) Windowed attention -> bf16.
    window_attn<<<M, 256, 0, stream>>>(qkv_f32, rel_b, attn_bf);

    // 5) proj GEMM + residual(x): -> x2 f32.
    gemm_bf16_wmma<1><<<dim3(C / 256, M / 128), 256, 0, stream>>>(
        attn_bf, projw_bf, proj_b, x, x2_f32, M, C, C);

    // 6) LN2 -> bf16.
    layernorm512_bf16<<<M, 128, 0, stream>>>(x2_f32, norm2_w, norm2_b, h_bf);

    // 7) FC1 GEMM + GELU -> bf16.
    gemm_bf16_wmma<2><<<dim3(HID / 256, M / 128), 256, 0, stream>>>(
        h_bf, fc1w_bf, fc1_b, nullptr, m_bf, M, HID, C);

    // 8) FC2 GEMM + residual(x2) -> final f32 output.
    gemm_bf16_wmma<1><<<dim3(C / 256, M / 128), 256, 0, stream>>>(
        m_bf, fc2w_bf, fc2_b, x2_f32, out, M, C, HID);
}
